// FeedForwardKernel_85615878078685
// MI455X (gfx1250) — matrix operand order repaired and checked by reading, not compiled, not run on hardware
//
#include <hip/hip_runtime.h>
#include <hip/hip_bf16.h>
#include <stdint.h>

// Problem constants (from reference setup_inputs)
#define BB   8
#define LQ   128
#define LK   512
#define HH   512
#define NEG_INF (-1e30f)
#define NQ   4      // q-values per block (key reuse factor)

typedef __attribute__((ext_vector_type(16))) _Float16 v16h;
typedef __attribute__((ext_vector_type(8)))  float    v8f;

__device__ __forceinline__ float fast_tanh(float x) {
#if __has_builtin(__builtin_amdgcn_tanhf)
  return __builtin_amdgcn_tanhf(x);           // v_tanh_f32 (CDNA5 TRANS op)
#elif __has_builtin(__builtin_amdgcn_exp2f)
  // tanh(|x|) = (1-t)/(1+t), t = e^{-2|x|}; restore sign.
  float ax = __builtin_fabsf(x);
  float t  = __builtin_amdgcn_exp2f(ax * -2.885390081777926f); // -2*log2(e)
  float r  = __builtin_amdgcn_rcpf(1.0f + t);
  float y  = __builtin_fmaf(-2.0f * t, r, 1.0f);
  return __builtin_copysignf(y, x);
#else
  return tanhf(x);
#endif
}

__device__ __forceinline__ float fast_exp(float x) {
#if __has_builtin(__builtin_amdgcn_exp2f)
  return __builtin_amdgcn_exp2f(x * 1.4426950408889634f);      // v_exp_f32
#else
  return __expf(x);
#endif
}

// One block: one b, NQ consecutive q's. 256 threads = 8 waves.
// Each wave owns k-tiles of 16 rows; WMMA reduces over H in chunks of 32.
__global__ __launch_bounds__(256)
void bahdanau_attn_wmma_kernel(const float* __restrict__ key,      // [B, LK, H]
                               const float* __restrict__ query,    // [LQ, B, H]
                               const unsigned char* __restrict__ key_mask, // [B, LK]
                               const float* __restrict__ weights,  // [1, H]
                               float* __restrict__ out)            // [B, LQ, LK]
{
  __shared__ __align__(16) float    s_q[NQ][HH];       // query vectors (f32)
  __shared__ __align__(32) _Float16 s_w[HH];           // weights (f16)
  __shared__ __align__(16) float    s_scores[NQ][LK];  // raw scores
  __shared__ float s_red[16];

  const int tid  = threadIdx.x;
  const int wave = tid >> 5;
  const int lane = tid & 31;

  const int b  = blockIdx.x / (LQ / NQ);
  const int q0 = (blockIdx.x % (LQ / NQ)) * NQ;

  // Stage query rows + f16 weights into LDS.
  for (int i = tid; i < NQ * HH; i += 256) {
    int qq = i >> 9, h = i & (HH - 1);
    s_q[qq][h] = query[((size_t)(q0 + qq) * BB + b) * HH + h];
  }
  for (int i = tid; i < HH; i += 256) s_w[i] = (_Float16)weights[i];
  __syncthreads();

  const int laneHi = lane >> 4;      // 0: K={0..7,16..23}, 1: K={8..15,24..31}
  const int rowSel = lane & 15;      // A-matrix row (local k index)
  const int hoff   = laneHi * 8;

  for (int tile = wave; tile < LK / 16; tile += 8) {
    const int   krow = tile * 16 + rowSel;
    const float* kptr = key + ((size_t)b * LK + krow) * HH;
    if (tile + 8 < LK / 16) {
      const float* knext = key + ((size_t)b * LK + (tile + 8) * 16 + rowSel) * HH;
      __builtin_prefetch(knext, 0, 1);   // global_prefetch_b8
    }

    v8f acc[NQ];
#pragma unroll
    for (int nq = 0; nq < NQ; ++nq) acc[nq] = (v8f){};

    for (int hi = 0; hi < HH / 32; ++hi) {
      const int hbase = hi * 32 + hoff;

      // Key chunk for this lane's A rows: h = hbase+{0..7} and hbase+16+{0..7}
      float kv[16];
      *(float4*)&kv[0]  = *(const float4*)(kptr + hbase);
      *(float4*)&kv[4]  = *(const float4*)(kptr + hbase + 4);
      *(float4*)&kv[8]  = *(const float4*)(kptr + hbase + 16);
      *(float4*)&kv[12] = *(const float4*)(kptr + hbase + 20);

      // B = w replicated across all 16 columns (same K order as A):
      // lanes 0-15 hold K={0..7,16..23}, lanes 16-31 hold K={8..15,24..31}.
      typedef __attribute__((ext_vector_type(8))) _Float16 v8h_t;
      const v8h_t blo = *(const v8h_t*)&s_w[hi * 32 + hoff];
      const v8h_t bhi = *(const v8h_t*)&s_w[hi * 32 + 16 + hoff];
      const v16h bmat = __builtin_shufflevector(blo, bhi, 0,1,2,3,4,5,6,7,8,9,10,11,12,13,14,15);

#pragma unroll
      for (int nq = 0; nq < NQ; ++nq) {
        float qv[16];
        *(float4*)&qv[0]  = *(const float4*)&s_q[nq][hbase];
        *(float4*)&qv[4]  = *(const float4*)&s_q[nq][hbase + 4];
        *(float4*)&qv[8]  = *(const float4*)&s_q[nq][hbase + 16];
        *(float4*)&qv[12] = *(const float4*)&s_q[nq][hbase + 20];

        v16h a;
#pragma unroll
        for (int j = 0; j < 8; ++j) {
          a[j]     = (_Float16)fast_tanh(kv[j]     + qv[j]);
          a[8 + j] = (_Float16)fast_tanh(kv[8 + j] + qv[8 + j]);
        }
        acc[nq] = __builtin_amdgcn_wmma_f32_16x16x32_f16(
            false, a, false, bmat, (short)0, acc[nq], false, false);
      }
    }

    // D layout: VGPR r -> row r (lanes 0-15) / row 8+r (lanes 16-31); all
    // columns identical, so lanes 0 and 16 publish the 16 scores of this tile.
    if (rowSel == 0) {
#pragma unroll
      for (int nq = 0; nq < NQ; ++nq)
#pragma unroll
        for (int r = 0; r < 8; ++r)
          s_scores[nq][tile * 16 + laneHi * 8 + r] = acc[nq][r];
    }
  }
  __syncthreads();

  // Mask (hoisted: same k ownership for every nq).
  unsigned char mk[LK / 256];
#pragma unroll
  for (int i = 0; i < LK / 256; ++i) mk[i] = key_mask[b * LK + tid + i * 256];

  // Softmax over Lk for each of the NQ query rows.
  for (int nq = 0; nq < NQ; ++nq) {
    float sc[LK / 256];
    float lmax = -3.4e38f;
#pragma unroll
    for (int i = 0; i < LK / 256; ++i) {
      float s = s_scores[nq][tid + i * 256];
      if (mk[i]) s = NEG_INF;
      sc[i] = s;
      lmax = fmaxf(lmax, s);
    }
    for (int off = 16; off > 0; off >>= 1)
      lmax = fmaxf(lmax, __shfl_xor(lmax, off, 32));
    if (lane == 0) s_red[wave] = lmax;
    __syncthreads();
    float gmax = s_red[0];
#pragma unroll
    for (int i = 1; i < 8; ++i) gmax = fmaxf(gmax, s_red[i]);

    float e[LK / 256];
    float lsum = 0.0f;
#pragma unroll
    for (int i = 0; i < LK / 256; ++i) { e[i] = fast_exp(sc[i] - gmax); lsum += e[i]; }
    for (int off = 16; off > 0; off >>= 1)
      lsum += __shfl_xor(lsum, off, 32);
    if (lane == 0) s_red[8 + wave] = lsum;
    __syncthreads();
    float gsum = 0.0f;
#pragma unroll
    for (int i = 0; i < 8; ++i) gsum += s_red[8 + i];
    const float inv = 1.0f / gsum;

    float* orow = out + ((size_t)b * LQ + (q0 + nq)) * LK;
#pragma unroll
    for (int i = 0; i < LK / 256; ++i) orow[tid + i * 256] = e[i] * inv;
    __syncthreads();   // s_red reused next iteration
  }
}

extern "C" void kernel_launch(void* const* d_in, const int* in_sizes, int n_in,
                              void* d_out, int out_size, void* d_ws, size_t ws_size,
                              hipStream_t stream) {
  const float*         key      = (const float*)d_in[0];          // [8,512,512]
  const float*         query    = (const float*)d_in[1];          // [128,8,512]
  const unsigned char* key_mask = (const unsigned char*)d_in[2];  // [8,512] bool
  const float*         weights  = (const float*)d_in[3];          // [1,512]
  float*               out      = (float*)d_out;                  // [8,128,512]

  dim3 grid(BB * (LQ / NQ));   // 256 blocks
  dim3 block(256);             // 8 waves (wave32)
  hipLaunchKernelGGL(bahdanau_attn_wmma_kernel, grid, block, 0, stream,
                     key, query, key_mask, weights, out);
}
